// GenerateActionsBlock_57312043597846
// MI455X (gfx1250) — matrix operand order repaired and checked by reading, not compiled, not run on hardware
//
#include <hip/hip_runtime.h>
#include <hip/hip_bf16.h>
#include <math.h>

typedef __bf16 v16bf __attribute__((ext_vector_type(16)));
typedef __bf16 v8bf  __attribute__((ext_vector_type(8)));
typedef float  v8f   __attribute__((ext_vector_type(8)));

constexpr int B_ = 16;
constexpr int N_ = 1024;
constexpr int E_ = 128;
constexpr int KV_STRIDE = 2 * E_ + 1;          // 257 floats per kv row
constexpr float RSQRT_E = 0.08838834764831845f; // 1/sqrt(128)

// ---------------------------------------------------------------------------
// Kernel 1: repack kv f32[row][257] -> bf16[row][256] (k at +0, v at +128),
// folding atoms_mask in. One block per row; 256 threads = 256 elements.
// Fixes the odd stride (257) so the WMMA kernel gets aligned vector loads.
// ---------------------------------------------------------------------------
__global__ __launch_bounds__(256) void repack_kv_bf16(
    const float* __restrict__ kv, const float* __restrict__ mask,
    __bf16* __restrict__ kvp)
{
  const int row = blockIdx.x;            // b*N + n  (16384 rows)
  const int e   = threadIdx.x;           // 0..255
  const float m = mask[row];             // uniform per block -> scalar load
  const float x = kv[(size_t)row * KV_STRIDE + e] * m;
  kvp[((size_t)row << 8) + e] = (__bf16)x;
}

// ---------------------------------------------------------------------------
// Kernel 2: fused  rel = K·Vt/sqrt(E)  +  pairwise-direction contraction.
// One wave32 per 16-row i-tile. rel tiles live only in WMMA accumulators.
// ---------------------------------------------------------------------------
__global__ __launch_bounds__(128) void fused_actions(
    const __bf16* __restrict__ kvp, const float* __restrict__ pos,
    const float* __restrict__ mask, const float* __restrict__ ascale,
    float* __restrict__ out)
{
  const int lane = threadIdx.x & 31;
  const int wave = threadIdx.x >> 5;
  const int tile = blockIdx.x * 4 + wave;    // 0..1023
  if (tile >= B_ * (N_ / 16)) return;
  const int b  = tile >> 6;                  // tile / (N/16)
  const int i0 = (tile & 63) << 4;

  const int l15  = lane & 15;
  const int hi   = (lane >> 4) & 1;          // which 16-lane half
  const int base = hi << 3;                  // A chunk select: 0 or 8
  const int m0   = hi << 3;                  // C/D row offset: 0 or 8

  const size_t browofs = (size_t)b * N_;
  const float* posb  = pos  + browofs * 3;
  const float* maskb = mask + browofs;

  // ---- per-batch mask sum (wave-redundant; 32 loads + 5 shuffles)
  float msum = 0.f;
  for (int t = lane; t < N_; t += 32) msum += maskb[t];
  #pragma unroll
  for (int off = 16; off; off >>= 1) msum += __shfl_xor(msum, off, 32);
  const float inv_msum = __builtin_amdgcn_rcpf(msum);

  // ---- A tile (k_mu rows i0..i0+15), resident in the 16-bit A WMMA layout:
  // lanes 0-15 hold K = {0..7, 16..23}+32*ks ; lanes 16-31 K = {8..15, 24..31}
  const __bf16* arow = kvp + ((browofs + (size_t)(i0 + l15)) << 8);
  v16bf A[4];
  #pragma unroll
  for (int ks = 0; ks < 4; ++ks) {
    const __bf16* ap = arow + ks * 32 + base;
    v8bf c1 = *(const v8bf*)(ap);          // 16B aligned
    v8bf c2 = *(const v8bf*)(ap + 16);     // 16B aligned
    A[ks] = __builtin_shufflevector(c1, c2, 0,1,2,3,4,5,6,7,8,9,10,11,12,13,14,15);
  }

  // ---- this lane's 8 output rows: positions + mask, resident
  float pix[8], piy[8], piz[8], mi[8];
  #pragma unroll
  for (int r = 0; r < 8; ++r) {
    const int i = i0 + m0 + r;
    pix[r] = posb[i * 3 + 0];
    piy[r] = posb[i * 3 + 1];
    piz[r] = posb[i * 3 + 2];
    mi[r]  = maskb[i];
  }

  float ax[8] = {}, ay[8] = {}, az[8] = {};

  // ---- main loop over 16-wide j tiles
  for (int j0 = 0; j0 < N_; j0 += 16) {
    const int jr = j0 + l15;               // lane's column j
    // B layout: lane = column j; lanes 0-15 hold K {0..7, 16..23}, 16-31 hold K {8..15, 24..31}
    const __bf16* vrow = kvp + ((browofs + (size_t)jr) << 8) + E_ + (hi << 3);

    v8f c = {0.f, 0.f, 0.f, 0.f, 0.f, 0.f, 0.f, 0.f};
    #pragma unroll
    for (int ks = 0; ks < 4; ++ks) {
      v8bf b1 = *(const v8bf*)(vrow + ks * 32);        // 16B aligned
      v8bf b2 = *(const v8bf*)(vrow + ks * 32 + 16);   // 16B aligned
      v16bf bt = __builtin_shufflevector(b1, b2, 0,1,2,3,4,5,6,7,8,9,10,11,12,13,14,15);
      c = __builtin_amdgcn_wmma_f32_16x16x32_bf16(
              false, A[ks], false, bt, (short)0, c, false, false);
    }

    const float pjx = posb[jr * 3 + 0];
    const float pjy = posb[jr * 3 + 1];
    const float pjz = posb[jr * 3 + 2];

    // contract rel tile against unit directions; i==j gives d=0 -> 0 (no NaN)
    #pragma unroll
    for (int r = 0; r < 8; ++r) {
      const float s  = c[r] * RSQRT_E;            // rel[i0+m0+r][jr]
      const float dx = pix[r] - pjx;
      const float dy = piy[r] - pjy;
      const float dz = piz[r] - pjz;
      const float n2 = dx * dx + dy * dy + dz * dz;
      const float w  = s * __builtin_amdgcn_rcpf(__builtin_amdgcn_sqrtf(n2) + 1e-8f);
      ax[r] += dx * w;
      ay[r] += dy * w;
      az[r] += dz * w;
    }
  }

  // ---- reduce each row's 3-vector across the 16 lanes that share it
  #pragma unroll
  for (int r = 0; r < 8; ++r) {
    #pragma unroll
    for (int off = 8; off; off >>= 1) {
      ax[r] += __shfl_xor(ax[r], off, 16);
      ay[r] += __shfl_xor(ay[r], off, 16);
      az[r] += __shfl_xor(az[r], off, 16);
    }
  }

  // ---- lanes 0 and 16 finalize + store their 8 rows
  if (l15 == 0) {
    const float as = ascale[0];
    #pragma unroll
    for (int r = 0; r < 8; ++r) {
      const int i = i0 + m0 + r;
      const float g = as * mi[r];
      float* o = out + (browofs + (size_t)i) * 3;
      o[0] = tanhf(ax[r] * inv_msum) * g;
      o[1] = tanhf(ay[r] * inv_msum) * g;
      o[2] = tanhf(az[r] * inv_msum) * g;
    }
  }
}

// ---------------------------------------------------------------------------
extern "C" void kernel_launch(void* const* d_in, const int* in_sizes, int n_in,
                              void* d_out, int out_size, void* d_ws, size_t ws_size,
                              hipStream_t stream) {
  const float* kv     = (const float*)d_in[0];   // (16,1024,257) f32
  const float* pos    = (const float*)d_in[1];   // (16,1024,3)   f32
  const float* mask   = (const float*)d_in[2];   // (16,1024)     f32
  const float* ascale = (const float*)d_in[3];   // (1,)          f32
  float* out = (float*)d_out;                    // (16,1024,3)   f32
  __bf16* kvp = (__bf16*)d_ws;                   // 16*1024*256 bf16 = 8 MB

  // 1) repack + mask-fold: one block per kv row
  repack_kv_bf16<<<B_ * N_, 256, 0, stream>>>(kv, mask, kvp);

  // 2) fused WMMA + pairwise-direction contraction:
  //    1024 i-tiles, 4 waves (128 threads) per block
  fused_actions<<<(B_ * (N_ / 16)) / 4, 128, 0, stream>>>(kvp, pos, mask, ascale, out);
}
